// Harmonic_attn_2336462209708
// MI455X (gfx1250) — matrix operand order repaired and checked by reading, not compiled, not run on hardware
//
#include <hip/hip_runtime.h>
#include <hip/hip_bf16.h>

typedef _Float16 half_t;
typedef __attribute__((ext_vector_type(16))) _Float16 v16h;
typedef __attribute__((ext_vector_type(8)))  _Float16 v8h;
typedef __attribute__((ext_vector_type(4)))  _Float16 v4h;
typedef __attribute__((ext_vector_type(8)))  float    v8f;

// Problem constants
#define BB   4
#define CC   64
#define TT   512
#define FF   100
#define MCC  128
#define NC   360

// Packed-fragment half offsets inside workspace (contiguous from 0)
#define OWV   0         // Wv  as A-frags: MT=8,  KT=6  -> 48*512
#define OWK   24576     // Wk  as A-frags: MT=8,  KT=6
#define OWKQ  49152     // Wkq as A-frags: MT=8,  KT=12 -> 96*512
#define OWO   98304     // Wo  as A-frags: MT=4,  KT=12 -> 48*512
#define OB1   122880    // h_mat^T as B-frags: KT=4 (K=f pad128), NT=23 (n pad368)
#define OB2   169984    // h_mat   as B-frags: KT=12 (K=n pad384), NT=7 (f pad112)
#define OPEND 212992
#define OVBUF 262144                       // v   f16 [B,MC,T,F]
#define OKBUF (262144 + 26214400)          // k/h f16 [B,MC,T,F]

static __device__ __forceinline__ v8f wmma_f16(v16h a, v16h b, v8f c) {
  return __builtin_amdgcn_wmma_f32_16x16x32_f16(false, a, false, b, (short)0, c,
                                                false, false);
}

// A-fragment element -> K offset inside a 32-wide K tile (contiguous order used by this file).
// Per lane this is one contiguous 16-half run: k0 = 16*hi.
static __device__ __forceinline__ int a_klocal(int e, int hi) {
  return e + (hi << 4);
}

static __device__ __forceinline__ v16h cat8(v8h lo, v8h hi) {
  return __builtin_shufflevector(lo, hi, 0, 1, 2, 3, 4, 5, 6, 7, 8, 9, 10, 11,
                                 12, 13, 14, 15);
}

// ---------------------------------------------------------------------------
// k0: pack weights / h_mat into fragment-major f16 buffers.
// Conv weights use the reordered K dimension k' = d*Cin + c so that the
// activation-side B fragments are contiguous in LDS.
// ---------------------------------------------------------------------------
static __device__ __forceinline__ half_t packA_conv(const float* W, int rel,
                                                    int KT, int Cin) {
  int fi = rel >> 9, r = rel & 511, lane = r >> 4, e = r & 15;
  int mt = fi / KT, kt = fi - mt * KT;
  int kl = a_klocal(e, lane >> 4);
  int m = mt * 16 + (lane & 15);
  int k = kt * 32 + kl;           // reordered: k = d*Cin + c
  int d = k / Cin, c = k - d * Cin;
  return (half_t)W[(m * Cin + c) * 3 + d];
}

__global__ __launch_bounds__(256) void k0_prep(const float* __restrict__ Wv,
                                               const float* __restrict__ Wk,
                                               const float* __restrict__ Wkq,
                                               const float* __restrict__ Wo,
                                               const float* __restrict__ hmat,
                                               half_t* __restrict__ ws) {
  int idx = blockIdx.x * 256 + threadIdx.x;
  if (idx >= OPEND) return;
  half_t val;
  if (idx < OWK) {
    val = packA_conv(Wv, idx - OWV, 6, CC);
  } else if (idx < OWKQ) {
    val = packA_conv(Wk, idx - OWK, 6, CC);
  } else if (idx < OWO) {
    val = packA_conv(Wkq, idx - OWKQ, 12, MCC);
  } else if (idx < OB1) {
    val = packA_conv(Wo, idx - OWO, 12, MCC);
  } else if (idx < OB2) {
    int rel = idx - OB1;
    int fi = rel >> 9, r = rel & 511, lane = r >> 4, e = r & 15;
    int kt = fi / 23, nt = fi - kt * 23;
    int k = kt * 32 + (e & 7) + ((lane >> 4) << 3) + ((e >> 3) << 4);   // k = frequency bin f
    int n = nt * 16 + (lane & 15);              // n = pitch candidate
    val = (k < FF && n < NC) ? (half_t)hmat[n * FF + k] : (half_t)0.f;
  } else {
    int rel = idx - OB2;
    int fi = rel >> 9, r = rel & 511, lane = r >> 4, e = r & 15;
    int kt = fi / 7, nt = fi - kt * 7;
    int k = kt * 32 + (e & 7) + ((lane >> 4) << 3) + ((e >> 3) << 4);   // k = pitch candidate
    int f = nt * 16 + (lane & 15);              // n = frequency bin
    val = (k < NC && f < FF) ? (half_t)hmat[k * FF + f] : (half_t)0.f;
  }
  ws[idx] = val;
}

// ---------------------------------------------------------------------------
// k1: dytanh + conv1 (v) and dytanh(xn^2) + k_conv (k_out), per (b,t) slice.
// Activations staged transposed: XT[fcol][c] = xn[c][fcol-1].
// ---------------------------------------------------------------------------
__global__ __launch_bounds__(256) void k1_front(
    const float* __restrict__ x, const float* na, const float* na1,
    const float* nb, const float* nb1, const float* ma, const float* ma1,
    const float* mb, const float* mb1, const half_t* __restrict__ pWv,
    const half_t* __restrict__ pWk, const float* __restrict__ bv,
    const float* __restrict__ bk, half_t* __restrict__ vout,
    half_t* __restrict__ kout) {
  __shared__ __align__(32) half_t xn_s[114][CC];
  __shared__ __align__(32) half_t kn_s[114][CC];

  int bt = blockIdx.x;
  int b = bt >> 9, t = bt & 511;

  for (int i = threadIdx.x; i < 114 * CC; i += 256) {
    ((half_t*)xn_s)[i] = (half_t)0.f;
    ((half_t*)kn_s)[i] = (half_t)0.f;
  }
  __syncthreads();

  float A0 = *na, A1 = *na1, Ma = *ma, Ma1 = *ma1;
  for (int i = threadIdx.x; i < CC * FF; i += 256) {
    int c = i / FF, f = i - c * FF;
    float xv = x[(((size_t)(b * CC + c) * TT) + t) * FF + f];
    float xn = A1 * tanhf(A0 * xv + nb[c]) + nb1[c];
    float kn = Ma1 * tanhf(Ma * xn * xn + mb[c]) + mb1[c];
    xn_s[f + 1][c] = (half_t)xn;
    kn_s[f + 1][c] = (half_t)kn;
  }
  __syncthreads();

  if (bt + 1 < (int)gridDim.x) {  // prefetch next slice (global_prefetch_b8)
    int b2 = (bt + 1) >> 9, t2 = (bt + 1) & 511;
    __builtin_prefetch(&x[(((size_t)(b2 * CC) * TT) + t2) * FF], 0, 1);
  }

  int wave = threadIdx.x >> 5, lane = threadIdx.x & 31;
  int hi = lane >> 4, ln = lane & 15;

  for (int u = wave; u < 16; u += 8) {
    int sel = u >> 3, mt = u & 7;
    const half_t* pw = sel ? pWk : pWv;
    const float* bias = sel ? bk : bv;
    half_t(*src)[CC] = sel ? kn_s : xn_s;
    half_t* dst = sel ? kout : vout;

    // weight fragments are nt-invariant: load once per unit
    v16h aw[6];
#pragma unroll
    for (int kt = 0; kt < 6; ++kt)
      aw[kt] = *(const v16h*)(pw + ((((mt * 6 + kt) * 32) + lane) << 4));

    for (int nt = 0; nt < 7; ++nt) {
      v8f acc = {0.f, 0.f, 0.f, 0.f, 0.f, 0.f, 0.f, 0.f};
#pragma unroll
      for (int kt = 0; kt < 6; ++kt) {
        int k0 = kt * 32 + (hi << 4);
        int d = k0 >> 6, c0 = k0 & 63;
        v16h bf = *(const v16h*)&src[nt * 16 + ln + d][c0];
        acc = wmma_f16(aw[kt], bf, acc);
      }
#pragma unroll
      for (int g = 0; g < 8; ++g) {
        int m = mt * 16 + g + (hi << 3);
        int f = nt * 16 + ln;
        if (f < FF)
          dst[(((size_t)(b * MCC + m) * TT) + t) * FF + f] =
              (half_t)(acc[g] + bias[m]);
      }
    }
  }
}

// ---------------------------------------------------------------------------
// k2: scores = rows @ h_mat^T, softmax over 360, h = kq @ h_mat (in-place).
// One wave per 16-row tile; rows padded to 128 halves; kq lives only in LDS.
// ---------------------------------------------------------------------------
__global__ __launch_bounds__(64) void k2_attn(half_t* __restrict__ kio,
                                              const half_t* __restrict__ pB1,
                                              const half_t* __restrict__ pB2) {
  __shared__ __align__(32) half_t rowbuf[2][16 * 128];  // rows, K zero-padded
  __shared__ __align__(32) half_t scbuf[2][16 * 384];   // scores / kq

  int wave = threadIdx.x >> 5, lane = threadIdx.x & 31;
  int hi = lane >> 4, ln = lane & 15;
  size_t tile = (size_t)blockIdx.x * 2 + wave;  // 16384 tiles total
  half_t* rows = kio + tile * 1600;
  half_t* rb = rowbuf[wave];
  half_t* sc = scbuf[wave];

  // zero the K pad (f in [100,128)), then stage 16 rows with 4-half chunks
  // (100 % 4 == 0 -> chunks never cross a row boundary).
  for (int i = lane; i < 16 * 7; i += 32) {
    int m = i / 7, q = i - m * 7;
    *(v4h*)(rb + m * 128 + FF + q * 4) = (v4h){0.f, 0.f, 0.f, 0.f};
  }
  for (int j = lane; j < 400; j += 32) {
    int m = j / 25, q = j - m * 25;
    *(v4h*)(rb + m * 128 + q * 4) = *(const v4h*)(rows + j * 4);
  }

  // build & cache the 4 A-fragments: two contiguous 8-half runs per lane
  v16h af[4];
#pragma unroll
  for (int kt = 0; kt < 4; ++kt) {
    int k0 = kt * 32 + (hi << 3);
    v8h lo = *(const v8h*)(rb + ln * 128 + k0);
    v8h hh = *(const v8h*)(rb + ln * 128 + k0 + 16);
    af[kt] = cat8(lo, hh);
  }

  // GEMM1: scores[16, 368]
  for (int nt = 0; nt < 23; ++nt) {
    v8f acc = {0.f, 0.f, 0.f, 0.f, 0.f, 0.f, 0.f, 0.f};
#pragma unroll
    for (int kt = 0; kt < 4; ++kt) {
      v16h bf = *(const v16h*)(pB1 + ((((kt * 23 + nt) * 32) + lane) << 4));
      acc = wmma_f16(af[kt], bf, acc);
    }
#pragma unroll
    for (int g = 0; g < 8; ++g) {
      int m = g + (hi << 3), n = nt * 16 + ln;
      sc[m * 384 + n] = (half_t)acc[g];
    }
  }

  // softmax over n<360; lane (ln, hi) handles row ln, half hi (180 values)
  {
    int n0 = hi * 180;
    float mx = -1e30f;
    for (int n = n0; n < n0 + 180; ++n)
      mx = fmaxf(mx, (float)sc[ln * 384 + n]);
    mx = fmaxf(mx, __shfl_xor(mx, 16, 32));
    float s = 0.f;
    for (int n = n0; n < n0 + 180; ++n)
      s += __expf((float)sc[ln * 384 + n] - mx);
    s += __shfl_xor(s, 16, 32);
    float inv = 1.0f / s;
    for (int n = n0; n < n0 + 180; ++n)
      sc[ln * 384 + n] = (half_t)(__expf((float)sc[ln * 384 + n] - mx) * inv);
    if (hi)
      for (int n = NC; n < 384; ++n) sc[ln * 384 + n] = (half_t)0.f;
  }

  // GEMM2 A-fragments depend only on kt: hoist all 12 out of the nt loop
  v16h aq[12];
#pragma unroll
  for (int kt = 0; kt < 12; ++kt) {
    int k0 = kt * 32 + (hi << 3);
    v8h lo = *(const v8h*)(sc + ln * 384 + k0);
    v8h hh = *(const v8h*)(sc + ln * 384 + k0 + 16);
    aq[kt] = cat8(lo, hh);
  }

  // GEMM2: h[16, 112] = kq @ h_mat ; write back over the same rows
  for (int nt = 0; nt < 7; ++nt) {
    v8f acc = {0.f, 0.f, 0.f, 0.f, 0.f, 0.f, 0.f, 0.f};
#pragma unroll
    for (int kt = 0; kt < 12; ++kt) {
      v16h bf = *(const v16h*)(pB2 + ((((kt * 7 + nt) * 32) + lane) << 4));
      acc = wmma_f16(aq[kt], bf, acc);
    }
#pragma unroll
    for (int g = 0; g < 8; ++g) {
      int m = g + (hi << 3), f = nt * 16 + ln;
      if (f < FF) rows[m * FF + f] = (half_t)acc[g];
    }
  }
}

// ---------------------------------------------------------------------------
// k3: h -> kq_conv -> *v -> conv_out -> +residual, per (b,t) slice.
// Both h and hv staged transposed: XT[fcol][c].
// ---------------------------------------------------------------------------
__global__ __launch_bounds__(256) void k3_back(
    const half_t* __restrict__ hin, const half_t* __restrict__ vin,
    const half_t* __restrict__ pWkq, const float* __restrict__ bkq,
    const half_t* __restrict__ pWo, const float* __restrict__ bo,
    const float* __restrict__ x, float* __restrict__ out) {
  __shared__ __align__(32) half_t hs[114][MCC];
  __shared__ __align__(32) half_t hvs[114][MCC];

  int bt = blockIdx.x;
  int b = bt >> 9, t = bt & 511;

  for (int i = threadIdx.x; i < 114 * MCC; i += 256) {
    ((half_t*)hs)[i] = (half_t)0.f;
    ((half_t*)hvs)[i] = (half_t)0.f;
  }
  __syncthreads();
  for (int i = threadIdx.x; i < MCC * FF; i += 256) {
    int c = i / FF, f = i - c * FF;
    hs[f + 1][c] = hin[(((size_t)(b * MCC + c) * TT) + t) * FF + f];
  }
  __syncthreads();

  if (bt + 1 < (int)gridDim.x) {
    int b2 = (bt + 1) >> 9, t2 = (bt + 1) & 511;
    __builtin_prefetch(&hin[(((size_t)(b2 * MCC) * TT) + t2) * FF], 0, 1);
  }

  int wave = threadIdx.x >> 5, lane = threadIdx.x & 31;
  int hi = lane >> 4, ln = lane & 15;

  // phase 1: hh = conv1x3(h, Wkq) + bkq ; hv = hh * v  (wave w -> mtile w)
  {
    int mt = wave;
    v16h aw[12];
#pragma unroll
    for (int kt = 0; kt < 12; ++kt)
      aw[kt] = *(const v16h*)(pWkq + ((((mt * 12 + kt) * 32) + lane) << 4));

    for (int nt = 0; nt < 7; ++nt) {
      v8f acc = {0.f, 0.f, 0.f, 0.f, 0.f, 0.f, 0.f, 0.f};
#pragma unroll
      for (int kt = 0; kt < 12; ++kt) {
        int k0 = kt * 32 + (hi << 4);
        int d = k0 >> 7, c0 = k0 & 127;
        v16h bf = *(const v16h*)&hs[nt * 16 + ln + d][c0];
        acc = wmma_f16(aw[kt], bf, acc);
      }
#pragma unroll
      for (int g = 0; g < 8; ++g) {
        int m = mt * 16 + g + (hi << 3);
        int f = nt * 16 + ln;
        if (f < FF) {
          float vv = (float)vin[(((size_t)(b * MCC + m) * TT) + t) * FF + f];
          hvs[f + 1][m] = (half_t)((acc[g] + bkq[m]) * vv);
        }
      }
    }
  }
  __syncthreads();

  // phase 2: out = conv1x3(hv, Wo) + bo + residual.
  // Each wave owns one mt (waves 0-3: nt 0-3, waves 4-7: nt 4-6) so the 12
  // Wo fragments are loaded exactly once per wave.
  {
    int mt = wave & 3;
    int nt0 = (wave >> 2) ? 4 : 0;
    int nt1 = (wave >> 2) ? 7 : 4;
    v16h aw[12];
#pragma unroll
    for (int kt = 0; kt < 12; ++kt)
      aw[kt] = *(const v16h*)(pWo + ((((mt * 12 + kt) * 32) + lane) << 4));

    for (int nt = nt0; nt < nt1; ++nt) {
      v8f acc = {0.f, 0.f, 0.f, 0.f, 0.f, 0.f, 0.f, 0.f};
#pragma unroll
      for (int kt = 0; kt < 12; ++kt) {
        int k0 = kt * 32 + (hi << 4);
        int d = k0 >> 7, c0 = k0 & 127;
        v16h bf = *(const v16h*)&hvs[nt * 16 + ln + d][c0];
        acc = wmma_f16(aw[kt], bf, acc);
      }
#pragma unroll
      for (int g = 0; g < 8; ++g) {
        int m = mt * 16 + g + (hi << 3);
        int f = nt * 16 + ln;
        if (f < FF) {
          size_t oi = (((size_t)(b * CC + m) * TT) + t) * FF + f;
          out[oi] = acc[g] + bo[m] + x[oi];
        }
      }
    }
  }
}

// ---------------------------------------------------------------------------
extern "C" void kernel_launch(void* const* d_in, const int* in_sizes, int n_in,
                              void* d_out, int out_size, void* d_ws,
                              size_t ws_size, hipStream_t stream) {
  (void)in_sizes; (void)n_in; (void)out_size; (void)ws_size;
  const float* x    = (const float*)d_in[0];
  const float* na   = (const float*)d_in[1];
  const float* na1  = (const float*)d_in[2];
  const float* nb   = (const float*)d_in[3];
  const float* nb1  = (const float*)d_in[4];
  const float* ma   = (const float*)d_in[5];
  const float* ma1  = (const float*)d_in[6];
  const float* mb   = (const float*)d_in[7];
  const float* mb1  = (const float*)d_in[8];
  const float* Wv   = (const float*)d_in[9];
  const float* bv   = (const float*)d_in[10];
  const float* Wk   = (const float*)d_in[11];
  const float* bk   = (const float*)d_in[12];
  const float* Wkq  = (const float*)d_in[13];
  const float* bkq  = (const float*)d_in[14];
  const float* Wo   = (const float*)d_in[15];
  const float* bo   = (const float*)d_in[16];
  const float* hmat = (const float*)d_in[17];

  half_t* ws = (half_t*)d_ws;
  half_t* pWv  = ws + OWV;
  half_t* pWk  = ws + OWK;
  half_t* pWkq = ws + OWKQ;
  half_t* pWo  = ws + OWO;
  half_t* pB1  = ws + OB1;
  half_t* pB2  = ws + OB2;
  half_t* vbuf = ws + OVBUF;
  half_t* kbuf = ws + OKBUF;

  k0_prep<<<(OPEND + 255) / 256, 256, 0, stream>>>(Wv, Wk, Wkq, Wo, hmat, ws);
  k1_front<<<BB * TT, 256, 0, stream>>>(x, na, na1, nb, nb1, ma, ma1, mb, mb1,
                                        pWv, pWk, bv, bk, vbuf, kbuf);
  k2_attn<<<(BB * MCC * TT / 16) / 2, 64, 0, stream>>>(kbuf, pB1, pB2);
  k3_back<<<BB * TT, 256, 0, stream>>>(kbuf, vbuf, pWkq, bkq, pWo, bo, x,
                                       (float*)d_out);
}
